// Attention_68659347194065
// MI455X (gfx1250) — matrix operand order repaired and checked by reading, not compiled, not run on hardware
//
// MI455X (gfx1250, wave32) — coverage-attention fused kernels.
// Pipeline (all on `stream`, graph-capture safe):
//   1) prep_weights:  bf16(Ua), bf16(Wa), bf16(Wa^T)      (elementwise)
//   2) power_step x7: Wa^i = Wa^(i-1) @ Wa   via v_wmma_f32_16x16x32_bf16
//   3) attn_scores:   fused  pk = topics_k@Ua^T,  q_i = query@(Wa^T)^i,
//                     s = tanh(q+pk)·va + b, scaled by coverage.
//                     bf16 WMMA, f32 accumulate, XOR-swizzled LDS tiles,
//                     double-buffered async weight-slab staging (ASYNCcnt).
//   4) attn_out:      softmax over K=8, mt = sum_k alpha_k * topics_k.
// Workspace: Ua_bf 512K | WaT_bf 512K | W_bf[8] 4M | scores 512K  (~5.5 MB)

#include <hip/hip_runtime.h>

#define BB    16384
#define KTOP  8
#define HID   512
#define EMB   512
#define NS    32            // weight slab rows (output columns per slab)
#define NCH   64            // 16B chunks per 512-elem bf16 row

#if defined(__gfx1250__) && __has_builtin(__builtin_amdgcn_global_load_async_to_lds_b128)
#define HAVE_ASYNC 1
#else
#define HAVE_ASYNC 0
#endif

#define AS_GLOBAL __attribute__((address_space(1)))
#define AS_LDS    __attribute__((address_space(3)))

typedef __attribute__((ext_vector_type(16))) __bf16 v16bf;
typedef __attribute__((ext_vector_type(8)))  __bf16 v8bf;
typedef __attribute__((ext_vector_type(8)))  float  v8f;
typedef int v4i_vs __attribute__((vector_size(16)));   // matches async-builtin param

__device__ __forceinline__ v8f vzero8() {
  v8f z = {0.f,0.f,0.f,0.f,0.f,0.f,0.f,0.f};
  return z;
}
__device__ __forceinline__ v8f wmma_bf16(v16bf a, v16bf b, v8f c) {
  return __builtin_amdgcn_wmma_f32_16x16x32_bf16(false, a, false, b, (short)0, c, false, false);
}
__device__ __forceinline__ v16bf cat8(v8bf lo, v8bf hi) {
  return __builtin_shufflevector(lo, hi, 0,1,2,3,4,5,6,7,8,9,10,11,12,13,14,15);
}

// ---- un-swizzled fragment loads (global memory, power-step kernel) ----------
__device__ __forceinline__ v16bf load_fragA_g(const __bf16* base, int row, int kc, int sel8) {
  const __bf16* p = base + row * HID + kc + sel8;
  return cat8(*(const v8bf*)(p), *(const v8bf*)(p + 16));
}
__device__ __forceinline__ v16bf load_fragB_g(const __bf16* base, int row, int kc, int sel8) {
  const __bf16* p = base + row * HID + kc + sel8;
  return cat8(*(const v8bf*)(p), *(const v8bf*)(p + 16));
}

// ---- XOR-swizzled fragment loads (LDS tiles) --------------------------------
// Tile layout: row-major [rows][512] bf16, but the 16B chunk index within a row
// is XORed with (row & 15) -> 16 lanes reading 16 consecutive rows hit 16
// distinct bank groups (optimal 2 bank passes per ds_load_b128 wave access).
__device__ __forceinline__ v16bf fragA_swz(const __bf16* base, int row, int kc, int sel8) {
  const v8bf* r = (const v8bf*)(base + row * HID);
  const int c0 = (kc + sel8) >> 3, sw = row & 15;
  return cat8(r[c0 ^ sw], r[(c0 + 2) ^ sw]);      // elems k..k+7 and k+16..k+23
}
__device__ __forceinline__ v16bf fragB_swz(const __bf16* base, int row, int kc, int sel8) {
  const v8bf* r = (const v8bf*)(base + row * HID);
  const int c0 = (kc + sel8) >> 3, sw = row & 15;
  return cat8(r[c0 ^ sw], r[(c0 + 2) ^ sw]);      // elems k..k+7 and k+16..k+23
}

__device__ __forceinline__ void wait_async0() {
#if HAVE_ASYNC
#if __has_builtin(__builtin_amdgcn_s_wait_asynccnt)
  __builtin_amdgcn_s_wait_asynccnt(0);
#else
  asm volatile("s_wait_asynccnt 0x0" ::: "memory");
#endif
#endif
}

// stage one NS x 512 bf16 weight slab into swizzled LDS
__device__ __forceinline__ void stage_w_slab(int tid, const __bf16* gsrc, __bf16* lds) {
#if HAVE_ASYNC
  AS_GLOBAL char* g = (AS_GLOBAL char*)gsrc;     // builtin takes non-const int4 ptrs
  AS_LDS char* l = (AS_LDS char*)lds;
  for (int i = tid; i < NS * NCH; i += 256) {
    const int r = i >> 6, c = i & 63;
    __builtin_amdgcn_global_load_async_to_lds_b128(
        (AS_GLOBAL v4i_vs*)(g + (size_t)i * 16),
        (AS_LDS v4i_vs*)(l + ((size_t)r * NCH + (c ^ (r & 15))) * 16),
        0, 0);
  }
#else
  const uint4* g = (const uint4*)gsrc;
  uint4* l = (uint4*)lds;
  for (int i = tid; i < NS * NCH; i += 256) {
    const int r = i >> 6, c = i & 63;
    l[r * NCH + (c ^ (r & 15))] = g[i];
  }
#endif
}

// stage a 64 x 512 f32 tile -> bf16 swizzled LDS
__device__ __forceinline__ void stage_cvt_tile(int tid, const float* src, size_t rstride,
                                               __bf16* lds) {
  for (int i = tid; i < 64 * NCH; i += 256) {
    const int r = i >> 6, c = i & 63;
    const float4* p = (const float4*)(src + (size_t)r * rstride + c * 8);
    const float4 a = p[0], b = p[1];
    v8bf o;
    o[0]=(__bf16)a.x; o[1]=(__bf16)a.y; o[2]=(__bf16)a.z; o[3]=(__bf16)a.w;
    o[4]=(__bf16)b.x; o[5]=(__bf16)b.y; o[6]=(__bf16)b.z; o[7]=(__bf16)b.w;
    *(v8bf*)(lds + r * HID + ((c ^ (r & 15)) * 8)) = o;
  }
}

// ---- 1) elementwise prep: bf16 copies of Ua, Wa, Wa^T -----------------------
__global__ __launch_bounds__(256)
void prep_weights_kernel(const float* __restrict__ Ua, const float* __restrict__ Wa,
                         __bf16* __restrict__ Ua_bf, __bf16* __restrict__ W0_bf,
                         __bf16* __restrict__ WaT_bf) {
  int i = blockIdx.x * 256 + threadIdx.x;            // grid covers exactly 512*512
  Ua_bf[i] = (__bf16)Ua[i];
  float w = Wa[i];
  W0_bf[i] = (__bf16)w;
  WaT_bf[(i & (HID - 1)) * HID + (i >> 9)] = (__bf16)w;
}

// ---- 2) Wa^i chain: one 16x16 tile per wave, K-loop 512/32 ------------------
__global__ __launch_bounds__(128)
void power_step_kernel(const __bf16* __restrict__ Ain,   // Wa^(i-1), row-major bf16
                       const __bf16* __restrict__ WaT,   // Wa^T, row-major bf16
                       __bf16* __restrict__ Out) {       // Wa^i
  const int lane = threadIdx.x & 31;
  const int wave = threadIdx.x >> 5;
  const int tile = blockIdx.x * 4 + wave;              // 1024 tiles (32x32)
  const int tm = tile >> 5, tn = tile & 31;
  const int l15 = lane & 15;
  const int sel8 = (lane >= 16) ? 8 : 0, sel16 = (lane >= 16) ? 16 : 0;
  v8f acc = vzero8();
  const int ar = tm * 16 + l15;
  const int br = tn * 16 + l15;
#pragma unroll
  for (int kc = 0; kc < HID; kc += 32) {
    acc = wmma_bf16(load_fragA_g(Ain, ar, kc, sel8), load_fragB_g(WaT, br, kc, sel8), acc);
  }
  const int mrow = tm * 16 + ((lane >= 16) ? 8 : 0);   // D: VGPR j -> M=j / j+8
  const int ncol = tn * 16 + l15;
#pragma unroll
  for (int j = 0; j < 8; ++j)
    Out[(mrow + j) * HID + ncol] = (__bf16)acc[j];
}

// ---- 3) fused score kernel --------------------------------------------------
// Block: 256 thr = 8 waves = 4 row-stripes x 2 n-groups; 64 batch rows/block.
// Dynamic LDS 256KB: qA[64][512] | tA[64][512] | wbuf[2 bufs][Ua,Wk][32][512]
__global__ __launch_bounds__(256)
void attn_scores_kernel(const float* __restrict__ query,
                        const float* __restrict__ topics,
                        const float* __restrict__ coverage,
                        const __bf16* __restrict__ Ua_bf,
                        const __bf16* __restrict__ W_bf,   // 8 x [512x512]
                        const float* __restrict__ va,
                        const float* __restrict__ va_b,
                        float* __restrict__ scores) {
  extern __shared__ __bf16 smem[];
  __bf16* qA = smem;                  // 64*512
  __bf16* tA = smem + 64 * HID;       // 64*512
  __bf16* wB = smem + 2 * 64 * HID;   // 4 slabs of NS*512: [buf][mat]
  __shared__ float sred[2][64];

  const int tid  = threadIdx.x;
  const int lane = tid & 31;
  const int wave = tid >> 5;
  const int wrow = wave & 3;          // row stripe 0..3
  const int wn   = wave >> 2;         // n-group 0..1
  const int l15  = lane & 15;
  const int sel8  = (lane >= 16) ? 8 : 0;
  const int sel16 = (lane >= 16) ? 16 : 0;
  const int row0 = blockIdx.x * 64;
  const float vb = va_b[0];
  const int arow = wrow * 16 + l15;
  const int brow = wn * 16 + l15;

  // stage query tile -> bf16 swizzled LDS (once per block)
  stage_cvt_tile(tid, query + (size_t)row0 * HID, HID, qA);

  for (int k = 0; k < KTOP; ++k) {
    const __bf16* Wk = W_bf + (size_t)k * HID * HID;
    __syncthreads();                                   // tA/wB/sred reuse guard
    // stage topics[:,k,:] tile -> bf16 swizzled LDS
    stage_cvt_tile(tid, topics + ((size_t)row0 * KTOP + k) * HID, (size_t)KTOP * HID, tA);
    // prefetch slab 0 into buffer 0
    stage_w_slab(tid, Ua_bf, wB + 0 * NS * HID);
    stage_w_slab(tid, Wk,    wB + 1 * NS * HID);

    float spart[8] = {0.f,0.f,0.f,0.f,0.f,0.f,0.f,0.f};

    for (int slab = 0; slab < HID / NS; ++slab) {
      wait_async0();                                   // my staging complete
      __syncthreads();                                 // everyone's staging + prev compute
      const int cur = slab & 1, nxt = cur ^ 1;
      if (slab + 1 < HID / NS) {                       // overlap next staging with WMMA
        stage_w_slab(tid, Ua_bf + (size_t)(slab + 1) * NS * HID, wB + (nxt * 2 + 0) * NS * HID);
        stage_w_slab(tid, Wk    + (size_t)(slab + 1) * NS * HID, wB + (nxt * 2 + 1) * NS * HID);
      }
      const __bf16* wUa = wB + (cur * 2 + 0) * NS * HID;
      const __bf16* wWk = wB + (cur * 2 + 1) * NS * HID;

      v8f accQ = vzero8(), accP = vzero8();
#pragma unroll
      for (int kc = 0; kc < HID; kc += 32) {
        v16bf aq = fragA_swz(qA, arow, kc, sel8);
        v16bf at = fragA_swz(tA, arow, kc, sel8);
        accP = wmma_bf16(at, fragB_swz(wUa, brow, kc, sel8), accP);
        accQ = wmma_bf16(aq, fragB_swz(wWk, brow, kc, sel8), accQ);
      }
      // epilogue: tanh(q+pk) * va, partial dot over this slab's 16 columns
      const float vav = va[slab * NS + wn * 16 + l15];
#pragma unroll
      for (int j = 0; j < 8; ++j)
        spart[j] += tanhf(accQ[j] + accP[j]) * vav;
    }
    // reduce over the 16 lanes of each half (n dimension)
#pragma unroll
    for (int j = 0; j < 8; ++j) {
      float x = spart[j];
      x += __shfl_xor(x, 1, 16);
      x += __shfl_xor(x, 2, 16);
      x += __shfl_xor(x, 4, 16);
      x += __shfl_xor(x, 8, 16);
      spart[j] = x;
    }
    if (l15 == 0) {
      int rloc = wrow * 16 + ((lane >> 4) ? 8 : 0);
#pragma unroll
      for (int j = 0; j < 8; ++j) sred[wn][rloc + j] = spart[j];
    }
    __syncthreads();
    if (tid < 64) {                                    // combine the 2 n-groups
      int r = row0 + tid;
      float s = (sred[0][tid] + sred[1][tid] + vb) * coverage[(size_t)r * KTOP + k];
      scores[(size_t)r * KTOP + k] = s;
    }
  }
}

// ---- 4) softmax over K=8 + weighted topic sum (memory-bound) ----------------
__global__ __launch_bounds__(256)
void attn_out_kernel(const float* __restrict__ topics,
                     const float* __restrict__ scores,
                     float* __restrict__ mt, float* __restrict__ alphas) {
  const int lane = threadIdx.x & 31;
  const int wave = threadIdx.x >> 5;
  const int row  = blockIdx.x * 8 + wave;              // grid 2048 -> 16384 rows
  float s[KTOP]; float m = -1e30f;
#pragma unroll
  for (int k = 0; k < KTOP; ++k) { s[k] = scores[(size_t)row * KTOP + k]; m = fmaxf(m, s[k]); }
  float sum = 0.f;
#pragma unroll
  for (int k = 0; k < KTOP; ++k) { s[k] = __expf(s[k] - m); sum += s[k]; }
  float inv = 1.0f / sum;
#pragma unroll
  for (int k = 0; k < KTOP; ++k) s[k] *= inv;
  if (lane < KTOP) alphas[(size_t)row * KTOP + lane] = s[lane];
  const float* trow = topics + (size_t)row * KTOP * EMB;
  for (int e = lane; e < EMB; e += 32) {
    float acc = 0.f;
#pragma unroll
    for (int k = 0; k < KTOP; ++k) acc += s[k] * trow[k * EMB + e];
    mt[(size_t)row * EMB + e] = acc;
  }
}

extern "C" void kernel_launch(void* const* d_in, const int* in_sizes, int n_in,
                              void* d_out, int out_size, void* d_ws, size_t ws_size,
                              hipStream_t stream) {
  (void)in_sizes; (void)n_in; (void)out_size; (void)ws_size;
  const float* query    = (const float*)d_in[0];
  const float* topics   = (const float*)d_in[1];
  const float* coverage = (const float*)d_in[2];
  const float* Ua       = (const float*)d_in[3];
  const float* Wa       = (const float*)d_in[4];
  const float* va       = (const float*)d_in[5];
  const float* va_b     = (const float*)d_in[6];

  float* mt     = (float*)d_out;
  float* alphas = mt + (size_t)BB * EMB;

  char* ws = (char*)d_ws;
  __bf16* Ua_bf  = (__bf16*)(ws);
  __bf16* WaT_bf = (__bf16*)(ws + 512 * 1024);
  __bf16* W_bf   = (__bf16*)(ws + 1024 * 1024);            // 8 x 512KB
  float*  scores = (float*)(ws + 5 * 1024 * 1024);         // B x 8

  prep_weights_kernel<<<HID * HID / 256, 256, 0, stream>>>(Ua, Wa, Ua_bf, W_bf, WaT_bf);
  for (int i = 1; i < KTOP; ++i)
    power_step_kernel<<<256, 128, 0, stream>>>(W_bf + (size_t)(i - 1) * HID * HID, WaT_bf,
                                               W_bf + (size_t)i * HID * HID);
  // LDS: qA 64K + tA 64K + 2 x (Ua,Wk) 32K slabs = 256KB
  attn_scores_kernel<<<BB / 64, 256, (2 * 64 * HID + 4 * NS * HID) * sizeof(__bf16), stream>>>(
      query, topics, coverage, Ua_bf, W_bf, va, va_b, scores);
  attn_out_kernel<<<BB / 8, 256, 0, stream>>>(topics, scores, mt, alphas);
}
